// PCELayer_68161130988044
// MI455X (gfx1250) — compile-verified, matrix operand order checked by reading, not run on hardware
//
#include <hip/hip_runtime.h>
#include <hip/hip_bf16.h>

// ---------------------------------------------------------------------------
// MoE patch-gated conv for MI455X (gfx1250, wave32, WMMA).
//
// Shapes: x[16,64,128,128] f32, expert_w[8,64,64,3,3], expert_b[8,64],
//         gate_w[8,6144], gate_b[8].  out[16,64,128,128] f32.
//
// Strategy (compute-bound, 155 GFLOP conv):
//   * implicit-GEMM conv via v_wmma_f32_16x16x32_bf16 (bf16 in / f32 acc)
//   * one workgroup = one 8x8 pixel block (== one gate patch), all 512
//     expert channels; one wave per expert
//   * x halo tile staged to LDS once (covers all K), weights pre-swizzled
//     into WMMA-B fragment layout in workspace (stays resident in L2)
//   * fused bias+relu+gate, cross-expert reduction via ds_add_f32, so the
//     512-channel intermediate never reaches HBM
// ---------------------------------------------------------------------------

typedef __attribute__((ext_vector_type(16))) __bf16 v16bf;
typedef __attribute__((ext_vector_type(8)))  float  v8f;

union BF16Frag {
    v16bf v;
    uint4 u[2];   // two 16B chunks = 8 VGPRs
};

// round-to-nearest-even f32 -> bf16 bits
static __device__ __forceinline__ unsigned short f2bf(float f) {
    union { float f; unsigned u; } x;
    x.f = f;
    unsigned u = x.u;
    return (unsigned short)((u + 0x7FFFu + ((u >> 16) & 1u)) >> 16);
}

// --------------------------------------------------------------------------
// Constants
// --------------------------------------------------------------------------
#define BATCH   16
#define CIN     64
#define HW      128
#define NEXP    8
#define OCPE    64          // out channels per expert
#define KTOT    576         // 64 channels * 9 taps
#define KSTEPS  18          // 576 / 32
#define NTILES  4           // 4 n-tiles of 16 per expert
#define WFRAG_ELEMS (NEXP * KSTEPS * NTILES * 32 * 16)   // 294912 bf16
#define WFRAG_BYTES (WFRAG_ELEMS * 2)                    // 589824 B

// --------------------------------------------------------------------------
// Kernel 1: swizzle expert_w into WMMA-B fragment layout (bf16).
// Fragment for (e, kstep s, ntile jn): lane l holds 16 bf16 contiguous:
//   elems 0..7  -> K = kb0..kb0+7,   elems 8..15 -> K = kb0+16..kb0+23
//   kb0 = (l>=16) ? 8 : 0 ;  column n = l%16 ;  global k = s*32 + K
//   k ordering: k = (kh*3+kw)*64 + c   (tap-major, channel-minor)
// --------------------------------------------------------------------------
__global__ __launch_bounds__(256) void prep_w_kernel(
        const float* __restrict__ ew, unsigned short* __restrict__ wfrag) {
    int t = blockIdx.x * 256 + threadIdx.x;
    if (t >= WFRAG_ELEMS) return;
    int i  = t & 15;
    int l  = (t >> 4) & 31;
    int jn = (t >> 9) & 3;
    int s  = (t >> 11) % KSTEPS;
    int e  = t / (KSTEPS * 2048);
    int oc = jn * 16 + (l & 15);
    int K  = (i & 7) + ((i & 8) ? 16 : 0) + ((l & 16) ? 8 : 0);
    int k  = s * 32 + K;
    int tap = k >> 6;            // kh*3+kw
    int c   = k & 63;
    float v = ew[(((e * OCPE + oc) * CIN + c) * 9) + tap];
    wfrag[t] = f2bf(v);
}

// --------------------------------------------------------------------------
// Kernel 2: per-patch gating (dot 6144 + softmax over 8 experts).
// One block per patch; wave w computes expert w's logit.
// --------------------------------------------------------------------------
__global__ __launch_bounds__(256) void gate_kernel(
        const float* __restrict__ x, const float* __restrict__ gate_w,
        const float* __restrict__ gate_b, float* __restrict__ gates) {
    int bid = blockIdx.x;                 // 0 .. 4095
    int b = bid >> 8;
    int p = bid & 255;
    int py = p >> 4, px = p & 15;
    int tid  = threadIdx.x;
    int lane = tid & 31;
    int e    = tid >> 5;                  // wave id == expert

    const float* gw = gate_w + e * 6144;
    float acc = 0.f;
    for (int f = lane; f < 6144; f += 32) {
        int c  = f >> 6;                  // 0..95 ("gate channel")
        int sp = f & 63;
        float gv;
        if (c < CIN) {
            int pp = sp >> 3, qq = sp & 7;
            gv = x[(((size_t)b * CIN + c) * HW + (py * 8 + pp)) * HW + (px * 8 + qq)];
        } else {
            int j = c - CIN;              // 0..31 fourier feature (const over p,q)
            int fr = j & 7;
            float coord = (((j >= 16) ? px : py) + 0.5f) * (1.0f / 16.0f);
            float ang = coord * 3.14159265358979f * (float)(1 << fr);
            gv = (j & 8) ? __cosf(ang) : __sinf(ang);
        }
        acc += gv * gw[f];
    }
    for (int o = 16; o > 0; o >>= 1) acc += __shfl_down(acc, o, 32);

    __shared__ float lg[NEXP];
    if (lane == 0) lg[e] = acc + gate_b[e];
    __syncthreads();

    if (tid == 0) {
        float mx = lg[0];
        #pragma unroll
        for (int k = 1; k < NEXP; ++k) mx = fmaxf(mx, lg[k]);
        float ex[NEXP], s = 0.f;
        #pragma unroll
        for (int k = 0; k < NEXP; ++k) { ex[k] = __expf(lg[k] - mx); s += ex[k]; }
        float inv = 1.0f / s;
        float* gout = gates + ((size_t)b * 256 + p) * NEXP;
        #pragma unroll
        for (int k = 0; k < NEXP; ++k) gout[k] = ex[k] * inv;
    }
}

// --------------------------------------------------------------------------
// Kernel 3: fused WMMA conv + bias + relu + gated expert reduction.
// Block = one 8x8 pixel block of one image (M=64), all 512 channels (N).
// Wave w = expert w: 4 M-tiles x 4 N-tiles accumulators (128 VGPRs).
// --------------------------------------------------------------------------
__global__ __launch_bounds__(256) void moe_conv_kernel(
        const float* __restrict__ x,
        const unsigned short* __restrict__ wfrag,
        const float* __restrict__ gates,
        const float* __restrict__ expert_b,
        float* __restrict__ out) {
    // LDS: halo x-tile 10x10x64 bf16 (covers all 18 K-steps), [y][x][c] order
    __shared__ __align__(16) unsigned short xt[10 * 10 * 64];   // 12.8 KB
    __shared__ float accs[64 * 64];                             // 16 KB f32

    int bid = blockIdx.x;
    int b = bid >> 8;
    int p = bid & 255;
    int h0 = (p >> 4) * 8, w0 = (p & 15) * 8;
    int tid = threadIdx.x;

    // ---- stage x halo tile (zero-padded SAME borders), f32 -> bf16 ----
    for (int idx = tid; idx < 6400; idx += 256) {
        int c   = idx & 63;
        int pos = idx >> 6;
        int yy  = pos / 10;
        int xx  = pos - yy * 10;
        int h = h0 - 1 + yy, w = w0 - 1 + xx;
        float v = 0.f;
        if ((unsigned)h < (unsigned)HW && (unsigned)w < (unsigned)HW)
            v = x[(((size_t)b * CIN + c) * HW + h) * HW + w];
        xt[idx] = f2bf(v);
    }
    for (int idx = tid; idx < 4096; idx += 256) accs[idx] = 0.f;
    __syncthreads();

    int lane = tid & 31;
    int e    = tid >> 5;         // wave == expert
    int lm   = lane & 15;
    int lh   = lane >> 4;        // 0/1 half of wave

    v8f zero = {0.f, 0.f, 0.f, 0.f, 0.f, 0.f, 0.f, 0.f};
    v8f acc[4][NTILES];
    #pragma unroll
    for (int mi = 0; mi < 4; ++mi)
        #pragma unroll
        for (int jn = 0; jn < NTILES; ++jn) acc[mi][jn] = zero;

    const unsigned short* wbase = wfrag + (size_t)e * KSTEPS * NTILES * 32 * 16;

    for (int s = 0; s < KSTEPS; ++s) {
        int tap = s >> 1;
        int dh = tap / 3, dw = tap - dh * 3;
        int cb = ((s & 1) << 5) + (lh << 3);     // channel base for this lane

        // A fragments from LDS: two contiguous ds_load_b128 per lane
        BF16Frag a[4];
        #pragma unroll
        for (int mi = 0; mi < 4; ++mi) {
            int m  = mi * 16 + lm;
            int py = m >> 3, px = m & 7;
            int off = (((py + dh) * 10 + (px + dw)) << 6) + cb;  // ushort idx
            a[mi].u[0] = *(const uint4*)&xt[off];
            a[mi].u[1] = *(const uint4*)&xt[off + 16];
        }
        // B fragments: coalesced global_load_b128 x2 (L2-resident)
        BF16Frag bb[NTILES];
        const unsigned short* wstep = wbase + ((size_t)(s * NTILES) * 32 + lane) * 16;
        #pragma unroll
        for (int jn = 0; jn < NTILES; ++jn) {
            const uint4* ptr = (const uint4*)(wstep + (size_t)jn * 32 * 16);
            bb[jn].u[0] = ptr[0];
            bb[jn].u[1] = ptr[1];
        }
        #pragma unroll
        for (int mi = 0; mi < 4; ++mi)
            #pragma unroll
            for (int jn = 0; jn < NTILES; ++jn)
                acc[mi][jn] = __builtin_amdgcn_wmma_f32_16x16x32_bf16(
                    false, a[mi].v, false, bb[jn].v,
                    (short)0, acc[mi][jn], false, false);
    }

    // ---- epilogue: bias + relu + gate scale, reduce experts via LDS ----
    float g = gates[((size_t)b * 256 + p) * NEXP + e];
    #pragma unroll
    for (int jn = 0; jn < NTILES; ++jn) {
        int oc = jn * 16 + lm;               // C/D: column N = lane%16
        float bias = expert_b[e * OCPE + oc];
        #pragma unroll
        for (int mi = 0; mi < 4; ++mi) {
            #pragma unroll
            for (int v = 0; v < 8; ++v) {    // C/D: row M = v + 8*(lane/16)
                float r = fmaxf(acc[mi][jn][v] + bias, 0.f) * g;
                int m = mi * 16 + v + (lh << 3);
                atomicAdd(&accs[(m << 6) + oc], r);   // ds_add_f32
            }
        }
    }
    __syncthreads();

    // ---- write out[b, oc, h0:h0+8, w0:w0+8], coalesced b128 stores ----
    int oc   = tid >> 2;
    int quad = tid & 3;
    #pragma unroll
    for (int rr = 0; rr < 2; ++rr) {
        int r = quad * 2 + rr;
        float4 v0, v1;
        v0.x = accs[((r * 8 + 0) << 6) + oc];
        v0.y = accs[((r * 8 + 1) << 6) + oc];
        v0.z = accs[((r * 8 + 2) << 6) + oc];
        v0.w = accs[((r * 8 + 3) << 6) + oc];
        v1.x = accs[((r * 8 + 4) << 6) + oc];
        v1.y = accs[((r * 8 + 5) << 6) + oc];
        v1.z = accs[((r * 8 + 6) << 6) + oc];
        v1.w = accs[((r * 8 + 7) << 6) + oc];
        float* dst = out + (((size_t)(b * OCPE + oc) * HW) + (h0 + r)) * HW + w0;
        *(float4*)dst       = v0;
        *(float4*)(dst + 4) = v1;
    }
}

// --------------------------------------------------------------------------
extern "C" void kernel_launch(void* const* d_in, const int* in_sizes, int n_in,
                              void* d_out, int out_size, void* d_ws, size_t ws_size,
                              hipStream_t stream) {
    const float* x        = (const float*)d_in[0];
    const float* expert_w = (const float*)d_in[1];
    const float* expert_b = (const float*)d_in[2];
    const float* gate_w   = (const float*)d_in[3];
    const float* gate_b   = (const float*)d_in[4];
    float* out = (float*)d_out;

    unsigned short* wfrag = (unsigned short*)d_ws;                    // 589824 B
    float* gates = (float*)((char*)d_ws + WFRAG_BYTES);               // 131072 B

    prep_w_kernel<<<(WFRAG_ELEMS + 255) / 256, 256, 0, stream>>>(expert_w, wfrag);
    gate_kernel<<<BATCH * 256, 256, 0, stream>>>(x, gate_w, gate_b, gates);
    moe_conv_kernel<<<BATCH * 256, 256, 0, stream>>>(x, wfrag, gates, expert_b, out);
}
